// GraphAttention_52604759441487
// MI455X (gfx1250) — matrix operand order repaired and checked by reading, not compiled, not run on hardware
//
#include <hip/hip_runtime.h>
#include <math.h>

typedef __attribute__((ext_vector_type(16))) __bf16 v16bf;
typedef __attribute__((ext_vector_type(8)))  float  v8f;

#define IN_DIM 256
#define HD     256   // H*D
#define NHEAD  4
#define DHEAD  64
#define NEG_SLOPE 0.2f

// ---------------------------------------------------------------------------
// Kernel W-pack: convert W [256,256] fp32 -> bf16 once (reused by all blocks)
// ---------------------------------------------------------------------------
__global__ void gat_wpack_kernel(const float* __restrict__ W,
                                 __bf16* __restrict__ Wbf)
{
    int idx = blockIdx.x * blockDim.x + threadIdx.x;
    if (idx < IN_DIM * HD) Wbf[idx] = (__bf16)W[idx];
}

// ---------------------------------------------------------------------------
// Kernel 1: ft = h @ W^T via v_wmma_f32_16x16x32_bf16 (wave32).
// Block = 128 threads = 4 waves; wave w owns one 64-wide head (4 N-tiles),
// reusing its A fragment across 4 WMMAs per K-chunk. K fully unrolled
// (8 chunks x 4 ntiles = 32 WMMAs per wave). B fragments load directly as
// bf16 (pre-packed) -> no cvt chain on the B path.
// ---------------------------------------------------------------------------
__global__ __launch_bounds__(128) void gat_gemm_kernel(
    const float* __restrict__ h, const __bf16* __restrict__ Wbf,
    float* __restrict__ ft, int n_nodes)
{
    const int lane = threadIdx.x & 31;
    const int wave = threadIdx.x >> 5;      // 0..3 -> head / 64-col group
    const int half = lane >> 4;             // 0: lanes 0-15, 1: lanes 16-31
    const int lm   = lane & 15;
    const int mtile = blockIdx.x;

    // A: lane holds row M = lm of the 16-row tile
    const float* __restrict__ arow = h + (size_t)(mtile * 16 + lm) * IN_DIM;
    // B = W^T: for ntile j, lane holds column N = lm -> row of W (bf16)
    const __bf16* __restrict__ brow0 = Wbf + (size_t)((wave * 4 + 0) * 16 + lm) * IN_DIM;
    const __bf16* __restrict__ brow1 = Wbf + (size_t)((wave * 4 + 1) * 16 + lm) * IN_DIM;
    const __bf16* __restrict__ brow2 = Wbf + (size_t)((wave * 4 + 2) * 16 + lm) * IN_DIM;
    const __bf16* __restrict__ brow3 = Wbf + (size_t)((wave * 4 + 3) * 16 + lm) * IN_DIM;

    v8f acc0 = {0.f,0.f,0.f,0.f,0.f,0.f,0.f,0.f};
    v8f acc1 = acc0, acc2 = acc0, acc3 = acc0;

#pragma unroll
    for (int kt = 0; kt < IN_DIM; kt += 32) {
        // A 16x32 bf16 layout: lanes 0-15 -> K {kt..kt+15}
        //                      lanes 16-31 -> K {kt+16..kt+31}
        const float* ap0 = arow + kt + 16 * half;       // elements 0..7
        const float* ap1 = arow + kt + 16 * half + 8;   // elements 8..15
        v16bf A;
#pragma unroll
        for (int e = 0; e < 8; ++e) {
            A[e]     = (__bf16)ap0[e];
            A[8 + e] = (__bf16)ap1[e];
        }

        // B 32x16 bf16 layout: element e -> K = kt + 16*half + e (contiguous,
        // 32-byte aligned -> direct v16bf load, no conversion)
        const int bo = kt + 16 * half;
        v16bf B0 = *(const v16bf*)(brow0 + bo);
        v16bf B1 = *(const v16bf*)(brow1 + bo);
        v16bf B2 = *(const v16bf*)(brow2 + bo);
        v16bf B3 = *(const v16bf*)(brow3 + bo);

        acc0 = __builtin_amdgcn_wmma_f32_16x16x32_bf16(
                   false, A, false, B0, (short)0, acc0, false, false);
        acc1 = __builtin_amdgcn_wmma_f32_16x16x32_bf16(
                   false, A, false, B1, (short)0, acc1, false, false);
        acc2 = __builtin_amdgcn_wmma_f32_16x16x32_bf16(
                   false, A, false, B2, (short)0, acc2, false, false);
        acc3 = __builtin_amdgcn_wmma_f32_16x16x32_bf16(
                   false, A, false, B3, (short)0, acc3, false, false);
    }

    // C/D 16x16 f32 layout: lane l, VGPR v -> M = v + 8*half, N = lm
    float* __restrict__ obase =
        ft + (size_t)(mtile * 16 + 8 * half) * HD + wave * 64 + lm;
#pragma unroll
    for (int v = 0; v < 8; ++v) obase[(size_t)v * HD +  0] = acc0[v];
#pragma unroll
    for (int v = 0; v < 8; ++v) obase[(size_t)v * HD + 16] = acc1[v];
#pragma unroll
    for (int v = 0; v < 8; ++v) obase[(size_t)v * HD + 32] = acc2[v];
#pragma unroll
    for (int v = 0; v < 8; ++v) obase[(size_t)v * HD + 48] = acc3[v];
}

// ---------------------------------------------------------------------------
// Kernel 0: init m = -inf, z = 0, out = 0 (harness poisons buffers)
// ---------------------------------------------------------------------------
__global__ void gat_init_kernel(float* __restrict__ out, float* __restrict__ m,
                                float* __restrict__ z, int n_nodes)
{
    int idx = blockIdx.x * blockDim.x + threadIdx.x;
    if (idx < n_nodes * HD) out[idx] = 0.f;
    if (idx < n_nodes * NHEAD) { m[idx] = -INFINITY; z[idx] = 0.f; }
}

// ---------------------------------------------------------------------------
// Kernel 2: per-node attention halves a1[n,h] = <ft[n,h,:], attn_l[h,:]>
// One block (256 thr) per node, LDS tree-reduce within each 64-wide head.
// ---------------------------------------------------------------------------
__global__ __launch_bounds__(256) void gat_attn_kernel(
    const float* __restrict__ ft, const float* __restrict__ attn_l,
    const float* __restrict__ attn_r, float* __restrict__ a1,
    float* __restrict__ a2)
{
    __shared__ float s1[256];
    __shared__ float s2[256];
    const int n = blockIdx.x;
    const int t = threadIdx.x;          // t = h*64 + d
    float f = ft[(size_t)n * HD + t];
    s1[t] = f * attn_l[t];
    s2[t] = f * attn_r[t];
    __syncthreads();
#pragma unroll
    for (int s = 32; s >= 1; s >>= 1) {
        if ((t & 63) < s) { s1[t] += s1[t + s]; s2[t] += s2[t + s]; }
        __syncthreads();
    }
    if ((t & 63) == 0) {
        int hh = t >> 6;
        a1[n * NHEAD + hh] = s1[t];
        a2[n * NHEAD + hh] = s2[t];
    }
}

// float atomic max via int/uint ordering trick (global_atomic_max_i32 / min_u32)
__device__ __forceinline__ void atomicMaxFloat(float* addr, float val)
{
    if (val >= 0.f)
        atomicMax((int*)addr, __float_as_int(val));
    else
        atomicMin((unsigned int*)addr, __float_as_uint(val));
}

// ---------------------------------------------------------------------------
// Kernel 3: edge logits = leaky_relu(a1[src]+a2[dst]); segment max into m
// One thread per (edge, head).
// ---------------------------------------------------------------------------
__global__ void gat_logits_kernel(const int* __restrict__ src,
                                  const int* __restrict__ dst,
                                  const float* __restrict__ a1,
                                  const float* __restrict__ a2,
                                  float* __restrict__ logits,
                                  float* __restrict__ m, int n_edges)
{
    int idx = blockIdx.x * blockDim.x + threadIdx.x;
    if (idx >= n_edges * NHEAD) return;
    int e  = idx >> 2;
    int hh = idx & 3;
    int s = src[e], d = dst[e];
    float x  = a1[s * NHEAD + hh] + a2[d * NHEAD + hh];
    float lg = (x > 0.f) ? x : NEG_SLOPE * x;
    logits[idx] = lg;
    atomicMaxFloat(&m[d * NHEAD + hh], lg);
}

// ---------------------------------------------------------------------------
// Kernel 4: a = exp(logit - m[dst]); out[dst,:] += ft[src,:]*a; z[dst] += a
// One block (256 thr) per edge; thread t handles feature t (head = t/64).
// ---------------------------------------------------------------------------
__global__ __launch_bounds__(256) void gat_scatter_kernel(
    const int* __restrict__ src, const int* __restrict__ dst,
    const float* __restrict__ ft, const float* __restrict__ logits,
    const float* __restrict__ m, float* __restrict__ out,
    float* __restrict__ z, int n_edges)
{
    int e = blockIdx.x;
    if (e >= n_edges) return;
    int t  = threadIdx.x;
    int hh = t >> 6;
    int s = src[e], d = dst[e];
    float a = expf(logits[e * NHEAD + hh] - m[d * NHEAD + hh]);
    atomicAdd(&out[(size_t)d * HD + t], ft[(size_t)s * HD + t] * a);
    if ((t & 63) == 0) atomicAdd(&z[d * NHEAD + hh], a);
}

// ---------------------------------------------------------------------------
// Kernel 5: out /= z (in place)
// ---------------------------------------------------------------------------
__global__ void gat_norm_kernel(float* __restrict__ out,
                                const float* __restrict__ z, int n_nodes)
{
    int idx = blockIdx.x * blockDim.x + threadIdx.x;
    if (idx >= n_nodes * HD) return;
    int n  = idx / HD;
    int hh = (idx & (HD - 1)) >> 6;
    out[idx] = out[idx] / z[n * NHEAD + hh];
}

// ---------------------------------------------------------------------------
extern "C" void kernel_launch(void* const* d_in, const int* in_sizes, int n_in,
                              void* d_out, int out_size, void* d_ws, size_t ws_size,
                              hipStream_t stream)
{
    const float* h      = (const float*)d_in[0];   // [N, 256]
    const float* W      = (const float*)d_in[1];   // [256, 256]
    const float* attn_l = (const float*)d_in[2];   // [4, 64]
    const float* attn_r = (const float*)d_in[3];   // [4, 64]
    const int*   src    = (const int*)d_in[4];     // [E]
    const int*   dst    = (const int*)d_in[5];     // [E]
    float* out = (float*)d_out;                    // [N, 4, 64]

    const int N = in_sizes[0] / IN_DIM;            // 50000
    const int E = in_sizes[4];                     // 800000

    // workspace carve-out
    float*  ft  = (float*)d_ws;                       // N*256 f32 (51.2 MB)
    __bf16* Wbf = (__bf16*)(ft + (size_t)N * HD);     // 256*256 bf16 (128 KB)
    float*  a1  = (float*)(Wbf + (size_t)IN_DIM * HD);// N*4
    float*  a2  = a1 + (size_t)N * NHEAD;             // N*4
    float*  m   = a2 + (size_t)N * NHEAD;             // N*4
    float*  z   = m  + (size_t)N * NHEAD;             // N*4
    float*  logits = z + (size_t)N * NHEAD;           // E*4

    // 0) init out/m/z + pack W to bf16
    gat_init_kernel<<<(N * HD + 255) / 256, 256, 0, stream>>>(out, m, z, N);
    gat_wpack_kernel<<<(IN_DIM * HD + 255) / 256, 256, 0, stream>>>(W, Wbf);
    // 1) GEMM via WMMA: 4 waves/block, each wave one head-wide 16x64 tile
    gat_gemm_kernel<<<N / 16, 128, 0, stream>>>(h, Wbf, ft, N);
    // 2) attention halves
    gat_attn_kernel<<<N, 256, 0, stream>>>(ft, attn_l, attn_r, a1, a2);
    // 3) edge logits + segment max
    gat_logits_kernel<<<(E * NHEAD + 255) / 256, 256, 0, stream>>>(
        src, dst, a1, a2, logits, m, E);
    // 4) softmax weights + scatter-sum
    gat_scatter_kernel<<<E, 256, 0, stream>>>(src, dst, ft, logits, m, out, z, E);
    // 5) normalize
    gat_norm_kernel<<<(N * HD + 255) / 256, 256, 0, stream>>>(out, z, N);
}
